// GET_86895778333059
// MI455X (gfx1250) — matrix operand order repaired and checked by reading, not compiled, not run on hardware
//
#include <hip/hip_runtime.h>
#include <math.h>

// ---------------- problem constants ----------------
#define N_  50000
#define E_  400000
#define G_  1024
#define D_  128
#define H_  4
#define DH_ 32
#define DE_ 16
#define HE_ 64
#define NRBF_ 16
#define L_  3
#define EPS_ 1e-5f
#define ISQ_ 0.17677669529663687f   // 1/sqrt(32)
#define NEGKEY_ 0x007FFFFFu         // order-preserving key of -inf

typedef __attribute__((ext_vector_type(16))) _Float16 v16h;
typedef __attribute__((ext_vector_type(8)))  _Float16 v8h;
typedef __attribute__((ext_vector_type(8)))  float    v8f;
typedef __attribute__((ext_vector_type(4)))  float    v4f;

// ---------------- helpers ----------------
__device__ __forceinline__ float wred(float v) {
    #pragma unroll
    for (int off = 16; off > 0; off >>= 1) v += __shfl_xor(v, off);
    return v;
}
__device__ __forceinline__ unsigned fkey(float f) {
    unsigned u = __float_as_uint(f);
    return (u & 0x80000000u) ? ~u : (u | 0x80000000u);
}
__device__ __forceinline__ float funkey(unsigned u) {
    unsigned b = (u & 0x80000000u) ? (u & 0x7FFFFFFFu) : ~u;
    return __uint_as_float(b);
}
// LDS byte offset of a pointer to __shared__ (LDS aperture keeps offset in addr[31:0])
__device__ __forceinline__ unsigned lds_off(const void* p) {
    return (unsigned)(size_t)p;
}
// async global->LDS 16B copy (gfx1250), tracked by ASYNCcnt
__device__ __forceinline__ void async_g2l_b128(unsigned lds, const void* gptr) {
    asm volatile("global_load_async_to_lds_b128 %0, %1, off"
                 :: "v"(lds), "v"(gptr) : "memory");
}
__device__ __forceinline__ void wait_async0() {
    asm volatile("s_wait_asynccnt 0" ::: "memory");
}

// ---------------- utility kernels ----------------
__global__ void fill_u32(unsigned* p, unsigned v, size_t n) {
    size_t i = (size_t)blockIdx.x * blockDim.x + threadIdx.x;
    if (i < n) p[i] = v;
}
__global__ void copy_f32(float* d, const float* s, size_t n) {
    size_t i = (size_t)blockIdx.x * blockDim.x + threadIdx.x;
    if (i < n) d[i] = s[i];
}
__global__ void count_f32(float* cnt, const int* ids, int n) {
    int i = blockIdx.x * blockDim.x + threadIdx.x;
    if (i < n) atomicAdd(&cnt[ids[i]], 1.0f);
}

// ---------------- weight pre-pack into WMMA fragment order ----------------
// dst layout: [K/32][Nc/16][lane=32][half=16], element =
//   W[ks*32 + (lane>>4)*16 + h][nb*16 + (lane&15)]   (zero for k >= Ksrc)
__global__ void pack_w(_Float16* dst, const float* __restrict__ W,
                       int Ksrc, int Kpad, int Nc)
{
    int i = blockIdx.x * blockDim.x + threadIdx.x;
    int total = Kpad * Nc;
    if (i >= total) return;
    int ncb  = Nc >> 4;
    int h    = i & 15;
    int lane = (i >> 4) & 31;
    int nb   = (i >> 9) % ncb;
    int ks   = (i >> 9) / ncb;
    int k = ks * 32 + (lane >> 4) * 16 + h;
    int c = nb * 16 + (lane & 15);
    float v = (k < Ksrc) ? W[(size_t)k * Nc + c] : 0.0f;
    dst[i] = (_Float16)v;
}

// ---------------- WMMA GEMM: out = act(A @ W + bias) + resid ----------------
// A: [M,K] (f32 or f16), Wp: packed f16 fragments, out/resid: [M,Nc]
// Requires K % 32 == 0, Nc % 64 == 0. Block tile 64x64, 8 waves,
// each wave: one 16-row A fragment reused by two WMMAs (16x32 output).
// Staging uses GLOBAL_LOAD_ASYNC_TO_LDS_B128 where no conversion is needed.
template<int AF16, int OF16>
__global__ __launch_bounds__(256) void gemm_wmma(
    const void* Av, const _Float16* __restrict__ Wp,
    const float* __restrict__ bias, const float* resid,
    void* Ov, int M, int K, int Nc, int act)
{
    __shared__ _Float16 Afrag[4][32][16];
    __shared__ _Float16 Bfrag[4][32][16];
    const int tid = threadIdx.x, lane = tid & 31, wave = tid >> 5;
    const int wr = wave & 3, wc = wave >> 2;
    const int row0 = blockIdx.y * 64, col0 = blockIdx.x * 64;

    // staging coordinates (one 16B chunk per thread per tile per k-step)
    const int blkS  = tid >> 6;          // A row-block / B n-block
    const int laneS = (tid >> 1) & 31;
    const int hgS   = tid & 1;
    int arow = row0 + blkS * 16 + (laneS & 15);
    if (arow >= M) arow = M - 1;         // clamp: masked rows only feed masked outputs
    const int akoff = (laneS >> 4) * 16 + hgS * 8;
    const int ncb = Nc >> 4;
    const size_t bsrc_base = ((size_t)(col0 >> 4) + blkS) * 512 + laneS * 16 + hgS * 8;
    const unsigned ldsA = lds_off(&Afrag[blkS][laneS][hgS * 8]);
    const unsigned ldsB = lds_off(&Bfrag[blkS][laneS][hgS * 8]);

    v8f acc0 = {}, acc1 = {};
    const int ksteps = K >> 5;
    for (int ks = 0; ks < ksteps; ++ks) {
        const int k0 = ks << 5;
        // stage A fragment chunk (8 halves, 16B)
        if (AF16) {
            async_g2l_b128(ldsA, (const _Float16*)Av + (size_t)arow * K + k0 + akoff);
        } else {
            const float* Ap = (const float*)Av + (size_t)arow * K + k0 + akoff;
            v4f f0 = *(const v4f*)Ap;
            v4f f1 = *(const v4f*)(Ap + 4);
            v8h ha;
            #pragma unroll
            for (int j = 0; j < 4; ++j) {
                ha[j]     = (_Float16)f0[j];
                ha[4 + j] = (_Float16)f1[j];
            }
            *(v8h*)&Afrag[blkS][laneS][hgS * 8] = ha;
        }
        // stage B fragment chunk (already packed f16, pure async 16B copy)
        async_g2l_b128(ldsB, Wp + (size_t)ks * ncb * 512 + bsrc_base);
        wait_async0();
        __syncthreads();

        v16h a  = *(const v16h*)&Afrag[wr][lane][0];
        v16h b0 = *(const v16h*)&Bfrag[wc * 2][lane][0];
        v16h b1 = *(const v16h*)&Bfrag[wc * 2 + 1][lane][0];
        acc0 = __builtin_amdgcn_wmma_f32_16x16x32_f16(
                   false, a, false, b0, (short)0, acc0, false, false);
        acc1 = __builtin_amdgcn_wmma_f32_16x16x32_f16(
                   false, a, false, b1, (short)0, acc1, false, false);
        __syncthreads();
    }
    // epilogue: D element (M = r + 8*(lane>=16), N = lane&15)
    const int nn  = lane & 15;
    const int mhi = (lane >> 4) * 8;
    #pragma unroll
    for (int r = 0; r < 8; ++r) {
        int gm = row0 + wr * 16 + r + mhi;
        if (gm >= M) continue;
        int gc = col0 + wc * 32 + nn;
        float v0 = acc0[r], v1 = acc1[r];
        if (bias) { v0 += bias[gc]; v1 += bias[gc + 16]; }
        if (act == 1) {
            v0 = v0 / (1.0f + __expf(-v0));
            v1 = v1 / (1.0f + __expf(-v1));
        }
        if (resid) {
            v0 += resid[(size_t)gm * Nc + gc];
            v1 += resid[(size_t)gm * Nc + gc + 16];
        }
        if (OF16) {
            ((_Float16*)Ov)[(size_t)gm * Nc + gc]      = (_Float16)v0;
            ((_Float16*)Ov)[(size_t)gm * Nc + gc + 16] = (_Float16)v1;
        } else {
            ((float*)Ov)[(size_t)gm * Nc + gc]      = v0;
            ((float*)Ov)[(size_t)gm * Nc + gc + 16] = v1;
        }
    }
}

// ---------------- row LayerNorm (wave per row, C in {64,128}) ----------------
__global__ __launch_bounds__(256) void ln_rows(
    float* x, const float* g, const float* b, int M, int C)
{
    int row = blockIdx.x * 8 + (threadIdx.x >> 5);
    if (row >= M) return;
    int lane = threadIdx.x & 31;
    int per = C >> 5;
    float vals[4]; float s = 0.0f;
    for (int i = 0; i < per; ++i) {
        vals[i] = x[(size_t)row * C + i * 32 + lane];
        s += vals[i];
    }
    float mean = wred(s) / (float)C;
    float vs = 0.0f;
    for (int i = 0; i < per; ++i) { float d = vals[i] - mean; vs += d * d; }
    float inv = rsqrtf(wred(vs) / (float)C + EPS_);
    for (int i = 0; i < per; ++i) {
        int c = i * 32 + lane;
        x[(size_t)row * C + c] = (vals[i] - mean) * inv * g[c] + b[c];
    }
}

// ---------------- edge segment-LN (per-graph per-channel) ----------------
__global__ void eseg_stats(const float* ea, const int* src, const int* batch,
                           float* esum, float* ecnt, int E)
{
    int e = blockIdx.x * blockDim.x + threadIdx.x;
    if (e >= E) return;
    int g = batch[src[e]];
    for (int c = 0; c < DE_; ++c) {
        float a = ea[(size_t)e * DE_ + c];
        atomicAdd(&esum[g * 32 + c], a);
        atomicAdd(&esum[g * 32 + 16 + c], a * a);
    }
    atomicAdd(&ecnt[g], 1.0f);
}
__global__ void eseg_apply(const float* ea, const int* src, const int* batch,
                           const float* esum, const float* ecnt,
                           const float* gn, const float* bn,
                           float* out, int ostride, size_t n)
{
    size_t i = (size_t)blockIdx.x * blockDim.x + threadIdx.x;
    if (i >= n) return;
    int e = (int)(i >> 4), c = (int)(i & 15);
    int g = batch[src[e]];
    float cnt = fmaxf(ecnt[g], 1.0f);
    float m = esum[g * 32 + c] / cnt;
    float var = esum[g * 32 + 16 + c] / cnt - m * m;
    out[(size_t)e * ostride + c] =
        (ea[i] - m) * rsqrtf(fmaxf(var, 0.0f) + EPS_) * gn[c] + bn[c];
}

// ---------------- per-graph position statistics ----------------
__global__ void pos_stats(const float* pos, const int* batch, float* seg,
                          int withsq, int n)
{
    int i = blockIdx.x * blockDim.x + threadIdx.x;
    if (i >= n) return;
    int g = batch[i];
    #pragma unroll
    for (int d = 0; d < 3; ++d) {
        float p = pos[i * 3 + d];
        atomicAdd(&seg[g * 6 + d], p);
        if (withsq) atomicAdd(&seg[g * 6 + 3 + d], p * p);
    }
}
__global__ void pos_scale(const float* seg, const float* ncnt, float* c,
                          float* scale, int withscale, int ng)
{
    int g = blockIdx.x * blockDim.x + threadIdx.x;
    if (g >= ng) return;
    float cnt = fmaxf(ncnt[g], 1.0f);
    #pragma unroll
    for (int d = 0; d < 3; ++d) {
        float m = seg[g * 6 + d] / cnt;
        c[g * 3 + d] = m;
        if (withscale) {
            float var = seg[g * 6 + 3 + d] / cnt - m * m;
            scale[g * 3 + d] = sqrtf(fmaxf(var, 0.0f) + EPS_);
        }
    }
}
__global__ void pos_apply_eqln(float* pos, const int* batch, const float* c,
                               const float* scale, int n3)
{
    int i = blockIdx.x * blockDim.x + threadIdx.x;
    if (i >= n3) return;
    int nidx = i / 3, d = i % 3, g = batch[nidx];
    float cc = c[g * 3 + d];
    pos[i] = cc + (pos[i] - cc) / scale[g * 3 + d];
}
__global__ void pos_gate_apply(float* pos, const int* batch, const float* c,
                               const float* gate, int n3)
{
    int i = blockIdx.x * blockDim.x + threadIdx.x;
    if (i >= n3) return;
    int nidx = i / 3, d = i % 3, g = batch[nidx];
    pos[i] += (pos[i] - c[g * 3 + d]) * gate[nidx];
}
__global__ void pos_msg_apply(float* pos, const float* posagg, const float* deg, int n3)
{
    int i = blockIdx.x * blockDim.x + threadIdx.x;
    if (i >= n3) return;
    pos[i] += posagg[i] / fmaxf(deg[i / 3], 1.0f);
}

// ---------------- attention: logits (wave per edge) ----------------
__global__ __launch_bounds__(256) void attn_logits(
    const float* q, const float* k, const float* ehid, const float* pos,
    const int* src, const int* dst, const float* Web, const float* Wrb,
    float* logit, unsigned* mmax, int E)
{
    int e = blockIdx.x * 8 + (threadIdx.x >> 5);
    if (e >= E) return;
    int lane = threadIdx.x & 31;
    int s = src[e], d = dst[e];
    float rx = pos[d * 3 + 0] - pos[s * 3 + 0];
    float ry = pos[d * 3 + 1] - pos[s * 3 + 1];
    float rz = pos[d * 3 + 2] - pos[s * 3 + 2];
    float dist = sqrtf(rx * rx + ry * ry + rz * rz + 1e-8f);

    float lg[4];
    float e0 = ehid[(size_t)e * HE_ + lane];
    float e1 = ehid[(size_t)e * HE_ + 32 + lane];
    #pragma unroll
    for (int h = 0; h < 4; ++h) {
        float p = q[(size_t)d * D_ + h * 32 + lane] *
                  k[(size_t)s * D_ + h * 32 + lane] * ISQ_;
        p += e0 * Web[lane * 4 + h] + e1 * Web[(lane + 32) * 4 + h];
        if (lane < NRBF_) {
            float mu = (7.0f / 15.0f) * (float)lane;
            float t = __expf(-(dist - mu) * (dist - mu));
            p += t * Wrb[lane * 4 + h];
        }
        lg[h] = wred(p);
    }
    if (lane < 4) {
        float Lg = lg[lane];
        logit[(size_t)e * 4 + lane] = Lg;
        atomicMax(&mmax[(size_t)d * 4 + lane], fkey(Lg));
    }
}

// ---------------- attention: exp + segment sum ----------------
__global__ void attn_expsum(float* logit, const unsigned* mmax, float* den,
                            const int* dst, size_t n)
{
    size_t i = (size_t)blockIdx.x * blockDim.x + threadIdx.x;
    if (i >= n) return;
    int e = (int)(i >> 2), h = (int)(i & 3);
    int d = dst[e];
    float ex = __expf(logit[i] - funkey(mmax[(size_t)d * 4 + h]));
    logit[i] = ex;
    atomicAdd(&den[(size_t)d * 4 + h], ex);
}

// ---------------- attention: messages (wave per edge) ----------------
__global__ __launch_bounds__(256) void attn_msg(
    const float* v, const float* logit, const float* den, const float* pos,
    const int* src, const int* dst, const float* Wc,
    _Float16* msg, float* xagg, float* posagg, int E)
{
    int e = blockIdx.x * 8 + (threadIdx.x >> 5);
    if (e >= E) return;
    int lane = threadIdx.x & 31;
    int s = src[e], d = dst[e];
    float rx = pos[d * 3 + 0] - pos[s * 3 + 0];
    float ry = pos[d * 3 + 1] - pos[s * 3 + 1];
    float rz = pos[d * 3 + 2] - pos[s * 3 + 2];
    float dist = sqrtf(rx * rx + ry * ry + rz * rz + 1e-8f);

    float wcsum = 0.0f;
    #pragma unroll
    for (int h = 0; h < 4; ++h) {
        int c = h * 32 + lane;
        float alpha = logit[(size_t)e * 4 + h] /
                      (den[(size_t)d * 4 + h] + 1e-9f);
        float mv = alpha * v[(size_t)s * D_ + c];
        msg[(size_t)e * D_ + c] = (_Float16)mv;
        atomicAdd(&xagg[(size_t)d * D_ + c], mv);
        wcsum += mv * Wc[c];
    }
    float g = tanhf(wred(wcsum));
    if (lane < 3) {
        float rel = pos[d * 3 + lane] - pos[s * 3 + lane];
        atomicAdd(&posagg[d * 3 + lane], g * rel / (dist + 1.0f));
    }
}

// ---------------- FFN gate: tanh(t @ Wp2) (wave per node) ----------------
__global__ __launch_bounds__(256) void gate_dot(const float* t, const float* Wp2,
                                                float* gate, int n)
{
    int nd = blockIdx.x * 8 + (threadIdx.x >> 5);
    if (nd >= n) return;
    int lane = threadIdx.x & 31;
    float s = 0.0f;
    #pragma unroll
    for (int j = 0; j < 4; ++j)
        s += t[(size_t)nd * D_ + j * 32 + lane] * Wp2[j * 32 + lane];
    s = wred(s);
    if (lane == 0) gate[nd] = tanhf(s);
}

// ---------------- pooling ----------------
__global__ void pool_acc(const float* x, const int* batch, unsigned* umax,
                         float* psum, size_t n)
{
    size_t i = (size_t)blockIdx.x * blockDim.x + threadIdx.x;
    if (i >= n) return;
    int nd = (int)(i >> 7), c = (int)(i & 127);
    int g = batch[nd];
    float v = x[i];
    atomicMax(&umax[(size_t)g * D_ + c], fkey(v));
    atomicAdd(&psum[(size_t)g * D_ + c], v);
}
__global__ void pool_write(const unsigned* umax, const float* psum,
                           const float* ncnt, float* out, int n)
{
    int i = blockIdx.x * blockDim.x + threadIdx.x;
    if (i >= n) return;
    int g = i >> 7, c = i & 127;
    out[(size_t)g * 256 + c] = funkey(umax[i]);
    out[(size_t)g * 256 + 128 + c] = psum[i] / fmaxf(ncnt[g], 1.0f);
}

// ================= host orchestration =================
extern "C" void kernel_launch(void* const* d_in, const int* in_sizes, int n_in,
                              void* d_out, int out_size, void* d_ws, size_t ws_size,
                              hipStream_t stream)
{
    (void)in_sizes; (void)n_in; (void)out_size; (void)ws_size;
    const float* x_in  = (const float*)d_in[0];
    const float* p_in  = (const float*)d_in[1];
    const float* ea_in = (const float*)d_in[2];
    auto P = [&](int i) { return (const float*)d_in[3 + i]; };
    const float* en_g = P(0);  const float* en_b = P(1);
    const float* el_W = P(2);  const float* el_b = P(3);
    const float* pre_gx = P(4); const float* pre_bx = P(5);
    const float* pre_ge = P(6); const float* pre_be = P(7);
    const float* Wq = P(8); const float* Wk = P(9); const float* Wv = P(10); const float* Wo = P(11);
    const float* Web = P(12); const float* Wrb = P(13); const float* Wc = P(14); const float* Weu = P(15);
    const float* W1 = P(16); const float* b1 = P(17); const float* W2 = P(18); const float* b2 = P(19);
    const float* Wp1 = P(20); const float* Wp2 = P(21); const float* We1 = P(22); const float* We2 = P(23);
    const float* ln0_gx = P(24); const float* ln0_bx = P(25);
    const float* ln0_ge = P(26); const float* ln0_be = P(27);
    const float* ln1_gx = P(28); const float* ln1_bx = P(29);
    const float* ln1_ge = P(30); const float* ln1_be = P(31);
    const int* batch = (const int*)d_in[35];
    const int* ei    = (const int*)d_in[36];
    const int* src = ei;
    const int* dst = ei + E_;

    // ---- workspace carve-out ----
    char* wsp = (char*)d_ws; size_t off = 0;
    auto take = [&](size_t bytes) -> char* {
        char* r = wsp + off; off = (off + bytes + 255) & ~(size_t)255; return r;
    };
    float*    xbuf   = (float*)take((size_t)N_ * D_ * 4);
    float*    posb   = (float*)take((size_t)N_ * 3 * 4);
    float*    ehid   = (float*)take((size_t)E_ * HE_ * 4);
    float*    qbuf   = (float*)take((size_t)N_ * D_ * 4);   // also h1 [N,256], etmp [E,32]
    float*    kbuf   = (float*)take((size_t)N_ * D_ * 4);
    float*    vbuf   = (float*)take((size_t)N_ * D_ * 4);
    float*    logit  = (float*)take((size_t)E_ * H_ * 4);
    _Float16* msg    = (_Float16*)take((size_t)E_ * D_ * 2); // also edge-FFN hidden (f16)
    float*    xagg   = (float*)take((size_t)N_ * D_ * 4);    // also Wp1 out
    float*    posagg = (float*)take((size_t)N_ * 3 * 4);
    unsigned* mmax   = (unsigned*)take((size_t)N_ * H_ * 4);
    float*    den    = (float*)take((size_t)N_ * H_ * 4);
    float*    deg    = (float*)take((size_t)N_ * 4);
    float*    ncnt   = (float*)take((size_t)G_ * 4);
    float*    ecnt   = (float*)take((size_t)G_ * 4);
    float*    esum   = (float*)take((size_t)G_ * 32 * 4);
    float*    seg6   = (float*)take((size_t)G_ * 6 * 4);
    float*    segc   = (float*)take((size_t)G_ * 3 * 4);
    float*    segs   = (float*)take((size_t)G_ * 3 * 4);
    float*    gate   = (float*)take((size_t)N_ * 4);
    unsigned* umax   = (unsigned*)take((size_t)G_ * D_ * 4);
    float*    psum   = (float*)take((size_t)G_ * D_ * 4);
    _Float16* wpack  = (_Float16*)take((size_t)256 * 128 * 2); // packed weight fragments
    float*    h1     = qbuf;   // [N,256] f32 (spans qbuf+kbuf, contiguous)
    float*    etmp   = qbuf;   // [E,32]  f32 zero-padded (spans qbuf+kbuf)

    auto fill = [&](void* p, unsigned v, size_t n) {
        fill_u32<<<(unsigned)((n + 255) / 256), 256, 0, stream>>>((unsigned*)p, v, n);
    };
    auto gemm = [&](const void* A, int af16, const float* W, int Ksrc,
                    const float* bias, const float* res, void* out, int of16,
                    int M, int K, int Nc, int act) {
        pack_w<<<(K * Nc + 255) / 256, 256, 0, stream>>>(wpack, W, Ksrc, K, Nc);
        dim3 g(Nc / 64, (M + 63) / 64);
        if (af16) {
            if (of16) gemm_wmma<1,1><<<g, 256, 0, stream>>>(A, wpack, bias, res, out, M, K, Nc, act);
            else      gemm_wmma<1,0><<<g, 256, 0, stream>>>(A, wpack, bias, res, out, M, K, Nc, act);
        } else {
            if (of16) gemm_wmma<0,1><<<g, 256, 0, stream>>>(A, wpack, bias, res, out, M, K, Nc, act);
            else      gemm_wmma<0,0><<<g, 256, 0, stream>>>(A, wpack, bias, res, out, M, K, Nc, act);
        }
    };
    auto posEqLn = [&]() {
        fill(seg6, 0, (size_t)G_ * 6);
        pos_stats<<<(N_ + 255) / 256, 256, 0, stream>>>(posb, batch, seg6, 1, N_);
        pos_scale<<<(G_ + 255) / 256, 256, 0, stream>>>(seg6, ncnt, segc, segs, 1, G_);
        pos_apply_eqln<<<(N_ * 3 + 255) / 256, 256, 0, stream>>>(posb, batch, segc, segs, N_ * 3);
    };

    // ---- init ----
    copy_f32<<<(unsigned)(((size_t)N_ * D_ + 255) / 256), 256, 0, stream>>>(xbuf, x_in, (size_t)N_ * D_);
    copy_f32<<<(unsigned)(((size_t)N_ * 3 + 255) / 256), 256, 0, stream>>>(posb, p_in, (size_t)N_ * 3);
    fill(ncnt, 0, G_); fill(deg, 0, N_); fill(ecnt, 0, G_);
    fill(esum, 0, (size_t)G_ * 32);
    count_f32<<<(N_ + 255) / 256, 256, 0, stream>>>(ncnt, batch, N_);
    count_f32<<<(E_ + 255) / 256, 256, 0, stream>>>(deg, dst, E_);

    // edge segment-LN + linear DE->HE (K padded 16 -> 32)
    eseg_stats<<<(E_ + 255) / 256, 256, 0, stream>>>(ea_in, src, batch, esum, ecnt, E_);
    fill(etmp, 0, (size_t)E_ * 32);
    eseg_apply<<<(unsigned)(((size_t)E_ * DE_ + 255) / 256), 256, 0, stream>>>(
        ea_in, src, batch, esum, ecnt, en_g, en_b, etmp, 32, (size_t)E_ * DE_);
    gemm(etmp, 0, el_W, DE_, el_b, nullptr, ehid, 0, E_, 32, HE_, 0);

    // pre EquivariantLayerNorm
    ln_rows<<<(N_ + 7) / 8, 256, 0, stream>>>(xbuf, pre_gx, pre_bx, N_, D_);
    ln_rows<<<(E_ + 7) / 8, 256, 0, stream>>>(ehid, pre_ge, pre_be, E_, HE_);
    posEqLn();

    for (int l = 0; l < L_; ++l) {
        // ---- attention ----
        gemm(xbuf, 0, Wq + (size_t)l * D_ * D_, D_, nullptr, nullptr, qbuf, 0, N_, D_, D_, 0);
        gemm(xbuf, 0, Wk + (size_t)l * D_ * D_, D_, nullptr, nullptr, kbuf, 0, N_, D_, D_, 0);
        gemm(xbuf, 0, Wv + (size_t)l * D_ * D_, D_, nullptr, nullptr, vbuf, 0, N_, D_, D_, 0);
        fill(mmax, NEGKEY_, (size_t)N_ * H_);
        fill(den, 0, (size_t)N_ * H_);
        attn_logits<<<(E_ + 7) / 8, 256, 0, stream>>>(
            qbuf, kbuf, ehid, posb, src, dst,
            Web + (size_t)l * HE_ * H_, Wrb + (size_t)l * NRBF_ * H_, logit, mmax, E_);
        attn_expsum<<<(unsigned)(((size_t)E_ * H_ + 255) / 256), 256, 0, stream>>>(
            logit, mmax, den, dst, (size_t)E_ * H_);
        fill(xagg, 0, (size_t)N_ * D_);
        fill(posagg, 0, (size_t)N_ * 3);
        attn_msg<<<(E_ + 7) / 8, 256, 0, stream>>>(
            vbuf, logit, den, posb, src, dst, Wc + (size_t)l * D_, msg, xagg, posagg, E_);
        gemm(xagg, 0, Wo + (size_t)l * D_ * D_, D_, nullptr, xbuf, xbuf, 0, N_, D_, D_, 0);
        pos_msg_apply<<<(N_ * 3 + 255) / 256, 256, 0, stream>>>(posb, posagg, deg, N_ * 3);
        gemm(msg, 1, Weu + (size_t)l * D_ * HE_, D_, nullptr, ehid, ehid, 0, E_, D_, HE_, 0);

        // ---- eq_ln (ln0) ----
        ln_rows<<<(N_ + 7) / 8, 256, 0, stream>>>(xbuf, ln0_gx + l * D_, ln0_bx + l * D_, N_, D_);
        ln_rows<<<(E_ + 7) / 8, 256, 0, stream>>>(ehid, ln0_ge + l * HE_, ln0_be + l * HE_, E_, HE_);
        posEqLn();

        // ---- FFN ----
        gemm(xbuf, 0, W1 + (size_t)l * D_ * 2 * D_, 2 * D_, b1 + (size_t)l * 2 * D_,
             nullptr, h1, 0, N_, D_, 2 * D_, 1);
        gemm(h1, 0, W2 + (size_t)l * 2 * D_ * D_, 2 * D_, b2 + (size_t)l * D_,
             xbuf, xbuf, 0, N_, 2 * D_, D_, 0);
        gemm(xbuf, 0, Wp1 + (size_t)l * D_ * D_, D_, nullptr, nullptr, xagg, 0, N_, D_, D_, 1);
        gate_dot<<<(N_ + 7) / 8, 256, 0, stream>>>(xagg, Wp2 + (size_t)l * D_, gate, N_);
        fill(seg6, 0, (size_t)G_ * 6);
        pos_stats<<<(N_ + 255) / 256, 256, 0, stream>>>(posb, batch, seg6, 0, N_);
        pos_scale<<<(G_ + 255) / 256, 256, 0, stream>>>(seg6, ncnt, segc, segs, 0, G_);
        pos_gate_apply<<<(N_ * 3 + 255) / 256, 256, 0, stream>>>(posb, batch, segc, gate, N_ * 3);
        gemm(ehid, 0, We1 + (size_t)l * HE_ * 2 * HE_, HE_, nullptr, nullptr,
             msg, 1, E_, HE_, 2 * HE_, 1);                       // f16 hidden into msg buffer
        gemm(msg, 1, We2 + (size_t)l * 2 * HE_ * HE_, 2 * HE_, nullptr, ehid, ehid, 0,
             E_, 2 * HE_, HE_, 0);

        // ---- eq_ln (ln1) ----
        ln_rows<<<(N_ + 7) / 8, 256, 0, stream>>>(xbuf, ln1_gx + l * D_, ln1_bx + l * D_, N_, D_);
        ln_rows<<<(E_ + 7) / 8, 256, 0, stream>>>(ehid, ln1_ge + l * HE_, ln1_be + l * HE_, E_, HE_);
        posEqLn();
    }

    // ---- final pooling (recover_scale on pos doesn't affect output; skipped) ----
    fill(umax, NEGKEY_, (size_t)G_ * D_);
    fill(psum, 0, (size_t)G_ * D_);
    pool_acc<<<(unsigned)(((size_t)N_ * D_ + 255) / 256), 256, 0, stream>>>(
        xbuf, batch, umax, psum, (size_t)N_ * D_);
    pool_write<<<(G_ * D_ + 255) / 256, 256, 0, stream>>>(umax, psum, ncnt, (float*)d_out, G_ * D_);
}
